// SpikingLayer_8770323218528
// MI455X (gfx1250) — matrix operand order repaired and checked by reading, not compiled, not run on hardware
//
#include <hip/hip_runtime.h>

typedef __attribute__((ext_vector_type(16))) __bf16       v16bf;
typedef __attribute__((ext_vector_type(8)))  float        v8f;
typedef __attribute__((ext_vector_type(8)))  unsigned int v8u;

#define IN_F   512
#define OUT_F  512
#define NIN    513          // IN_F + bias
#define BATCH  128
#define STEPS  200
#define TPAD   224          // padded time dim (multiple of 32, >= 13*16)
#define MT     13           // M tiles: 13*16 = 208 >= 200
#define OTILE  32
#define NT     (OTILE / 16) // 2
#define KTI    (TPAD / 32)  // 7 k-steps
#define PKLEN  432          // reversed pair table length (covers j in [0,430])
#define RES_F      1e-4f
#define RES_INV    10000.0f
#define SIM_TIME_F 2e-2f
// exp(-RES/TAU_SYN) = exp(-0.02)
#define ONE_M_DECAY 0.0198013266932447f
#define LOG_DECAY   (-0.02f)

__device__ __forceinline__ float kcoef(int d) {
    // K[d] = (1-a)(d+1)a^d for d >= 0, else 0
    float fd = (float)d;
    float c  = ONE_M_DECAY * (fd + 1.0f) * __expf(LOG_DECAY * fd);
    return (d >= 0) ? c : 0.0f;
}

__device__ __forceinline__ unsigned short bf16bits(float f) {
    return __builtin_bit_cast(unsigned short, (__bf16)f);
}

__global__ __launch_bounds__(256)
void snn_first_spike_kernel(const float* __restrict__ times,
                            const float* __restrict__ weights,
                            const float* __restrict__ delays,
                            const float* __restrict__ thresholds,
                            float* __restrict__ out)
{
    __shared__ float        s_charge[TPAD * OTILE]; // f32 charge; reused as mem after GEMM
    __shared__ __bf16       s_cbfT[OTILE * TPAD];   // bf16 charge, TRANSPOSED [o][s]
    __shared__ float        s_times[NIN];
    __shared__ unsigned int s_pk[PKLEN];            // packed {K(207-j), K(206-j)} pairs

    const int tid   = threadIdx.x;
    const int b     = blockIdx.x;
    const int obase = blockIdx.y * OTILE;

    // ---- phase 0: init LDS ----
    for (int j = tid; j < TPAD * OTILE; j += 256) s_charge[j] = 0.0f;
    for (int j = tid; j < NIN; j += 256)
        s_times[j] = (j < IN_F) ? times[b * IN_F + j] : 0.0f;  // bias spikes at t=0
    for (int j = tid; j < PKLEN; j += 256) {
        int d0 = 207 - j;
        unsigned int lo = bf16bits(kcoef(d0));
        unsigned int hi = bf16bits(kcoef(d0 - 1));
        s_pk[j] = lo | (hi << 16);
    }
    __syncthreads();

    // ---- phase 1: scatter synaptic charge into time bins (LDS float atomics) ----
    {
        const int ol = tid & (OTILE - 1);
        const int o  = obase + ol;
        for (int i = tid >> 5; i < NIN; i += 8) {
            float arr  = s_times[i] + delays[i * OUT_F + o];
            int   step = (int)rintf(arr * RES_INV);          // round-half-even == jnp.round
            step = min(max(step, 0), STEPS - 1);
            atomicAdd(&s_charge[step * OTILE + ol], weights[i * OUT_F + o]);
        }
    }
    __syncthreads();

    // ---- phase 2: f32 -> bf16, transposed so B fragments are contiguous pairs ----
    for (int j = tid; j < TPAD * OTILE; j += 256) {
        int o = j & (OTILE - 1);
        int s = j >> 5;
        s_cbfT[o * TPAD + s] = (__bf16)s_charge[s * OTILE + o];
    }
    __syncthreads();

    // ---- phase 3: mem[t,o] = sum_s K[t-s] * charge[s,o]  via v_wmma_f32_16x16x32_bf16 ----
    {
        const int wave = tid >> 5;
        const int lane = tid & 31;
        const int hl   = lane >> 4;   // lane half
        const int lrow = lane & 15;
        for (int tile = wave; tile < MT * NT; tile += 8) {     // wave-uniform loop
            const int mt = tile % MT;
            const int nt = tile / MT;
            const int t  = mt * 16 + lrow;                     // A row for this lane
            const int n  = nt * 16 + lrow;                     // B column for this lane
            const unsigned int* bT =
                (const unsigned int*)&s_cbfT[n * TPAD];        // pair-packed column
            v8f acc = {};
            for (int kk = 0; kk < KTI; ++kk) {
                // A fragment: VGPR v = {K(t-s), K(t-s-1)} = s_pk[207 - t + s]
                const int jb = 207 - t + kk * 32;
                v8u au, bu;
                #pragma unroll
                for (int v = 0; v < 8; ++v) {
                    int kl = (hl << 4) + (v << 1);
                    au[v] = s_pk[jb + kl];
                }
                // B fragment: VGPR v = {charge[s], charge[s+1]} packed in transposed column
                const int sb = kk * 16 + (hl << 3);
                #pragma unroll
                for (int v = 0; v < 8; ++v) {
                    bu[v] = bT[sb + v];
                }
                acc = __builtin_amdgcn_wmma_f32_16x16x32_bf16(
                        false, __builtin_bit_cast(v16bf, au),
                        false, __builtin_bit_cast(v16bf, bu),
                        (short)0, acc, false, false);
            }
            // D (16x16 f32 layout): lane holds N=lrow, rows hl*8 + r
            const int mbase = hl * 8;
            #pragma unroll
            for (int r = 0; r < 8; ++r) {
                int tt = mt * 16 + mbase + r;
                s_charge[tt * OTILE + n] = acc[r];   // reuse buffer as mem[t][o]
            }
        }
    }
    __syncthreads();

    // ---- phase 4: first threshold crossing + linear interpolation ----
    if (tid < OTILE) {
        const int   o   = obase + tid;
        const float thr = thresholds[o];
        float mem_prev = 0.0f;
        float spike    = SIM_TIME_F;     // "no spike" -> sim_time
        bool  spiked   = false;
        for (int t = 0; t < STEPS; ++t) {
            float m = s_charge[t * OTILE + tid];
            if (!spiked && m >= thr) {
                float denom = m - mem_prev;
                float safe  = (fabsf(denom) > 1e-12f) ? denom : 1e-12f;
                float frac  = (thr - mem_prev) / safe;
                frac  = fminf(fmaxf(frac, 0.0f), 1.0f);
                spike = ((float)t + frac) * RES_F;
                spiked = true;
            }
            mem_prev = m;
        }
        out[b * OUT_F + o] = spike;
    }
}

extern "C" void kernel_launch(void* const* d_in, const int* in_sizes, int n_in,
                              void* d_out, int out_size, void* d_ws, size_t ws_size,
                              hipStream_t stream) {
    (void)in_sizes; (void)n_in; (void)d_ws; (void)ws_size; (void)out_size;
    const float* times      = (const float*)d_in[0];
    const float* weights    = (const float*)d_in[1];
    const float* delays     = (const float*)d_in[2];
    const float* thresholds = (const float*)d_in[3];
    float* out = (float*)d_out;

    dim3 grid(BATCH, OUT_F / OTILE);   // 128 x 16 blocks
    snn_first_spike_kernel<<<grid, 256, 0, stream>>>(times, weights, delays, thresholds, out);
}
